// WrnYolov2_34282428957362
// MI455X (gfx1250) — matrix operand order repaired and checked by reading, not compiled, not run on hardware
//
#include <hip/hip_runtime.h>
#include <hip/hip_bf16.h>
#include <stdint.h>

// ---------------- types ----------------
typedef __bf16 bf16_t;
typedef bf16_t bf16x8  __attribute__((ext_vector_type(8)));
typedef bf16_t bf16x16 __attribute__((ext_vector_type(16)));
typedef float  f32x8   __attribute__((ext_vector_type(8)));
typedef unsigned int u32x4 __attribute__((ext_vector_type(4)));
typedef int    i32x4   __attribute__((ext_vector_type(4)));
typedef int    i32x8   __attribute__((ext_vector_type(8)));

// ---------------- geometry ----------------
#define BATCH 64
#define HIN   19
#define HP    21                 // padded 21x21
#define NPPI  (HP*HP)            // 441
#define NP    (BATCH*NPPI)       // 28224 padded pixel count
#define CCAT  1280
#define KTOT  (CCAT*9)           // 11520
#define C2    1024
#define M3    600
#define M3P   640

// ---------------- workspace layout (bytes) ----------------
#define OFF_CATP 0u                       // 1280*28224*2 = 72,253,440
#define OFF_PREP 72253440u                // 1024*28224*2 = 57,802,752
#define OFF_W1B  130056192u               // 64*512*2
#define OFF_W2B  130121728u               // 1024*11520*2 = 23,592,960
#define OFF_CWB  153714688u               // 640*1024*2
#define OFF_B1   155025408u               // 64*4
#define OFF_B2   155025664u               // 1024*4
#define OFF_CB   155029760u               // 640*4

#if defined(__has_builtin)
#  if __has_builtin(__builtin_amdgcn_tensor_load_to_lds)
#    define HAVE_TDM 1
#  endif
#  if __has_builtin(__builtin_amdgcn_s_wait_tensorcnt)
#    define HAVE_TCNT 1
#  endif
#endif

__device__ __forceinline__ float leaky(float y){ return y > 0.f ? y : 0.1f*y; }

// ---------------- WMMA fragment loaders (ISA VGPR layouts) ----------------
// A (16x32 bf16): lane<16 holds M=lane, K {0..7,16..23}; lane>=16 same M, K {8..15,24..31}
__device__ __forceinline__ bf16x16 load_a_frag(const bf16_t* row0, int rowStride, int lane){
  int m  = lane & 15;
  int ko = (lane >> 4) * 8;
  const bf16_t* r = row0 + m * rowStride;
  bf16x8 c0 = *(const bf16x8*)(r + ko);
  bf16x8 c1 = *(const bf16x8*)(r + 16 + ko);
  bf16x16 a;
#pragma unroll
  for (int i = 0; i < 8; ++i){ a[i] = c0[i]; a[i+8] = c1[i]; }
  return a;
}
// B (32x16 bf16) from [px][c] tile (c contiguous): lane column n=lane&15, K {0..7,16..23} (lane<16) / {8..15,24..31} (lane>=16)
__device__ __forceinline__ bf16x16 load_b_frag(const bf16_t* colbase, int rowStride, int lane){
  int n  = lane & 15;
  int ko = (lane >> 4) * 8;
  const bf16_t* r = colbase + n * rowStride + ko;
  bf16x8 c0 = *(const bf16x8*)(r);
  bf16x8 c1 = *(const bf16x8*)(r + 16);
  bf16x16 b;
#pragma unroll
  for (int i = 0; i < 8; ++i){ b[i] = c0[i]; b[i+8] = c1[i]; }
  return b;
}
__device__ __forceinline__ f32x8 wmma_bf16(bf16x16 a, bf16x16 b, f32x8 c){
  return __builtin_amdgcn_wmma_f32_16x16x32_bf16(false, a, false, b, (short)0, c, false, false);
}

// ---------------- TDM tile load (D# per CDNA5 ISA) ----------------
// Up-to-3D tile: x = t0 elems (contiguous, 2B), y = t1 rows (stride s0 elems),
// z = t2 planes (stride s1 elems). t2 == 0 -> plain 2D tile.
#if defined(HAVE_TDM)
__device__ __forceinline__ void tdm_load_tile(uint32_t lds_byte_off, const void* gp,
                                              uint32_t t0, uint32_t t1, uint32_t t2,
                                              uint64_t s0, uint64_t s1){
  uint64_t ga = (uint64_t)(uintptr_t)gp;
  u32x4 g0 = { 1u,                                   // count=1, user descriptor
               lds_byte_off,
               (uint32_t)ga,
               (uint32_t)(((ga >> 32) & 0x01FFFFFFull) | 0x80000000u) }; // addr[56:32] | type=2
  i32x8 g1;
  g1[0] = (int)(1u << 16);                                       // data_size=1 (2 bytes)
  g1[1] = (int)((t0 & 0xFFFFu) << 16);                           // tensor_dim0 = t0
  g1[2] = (int)(((t0 >> 16) & 0xFFFFu) | ((t1 & 0xFFFFu) << 16));// tensor_dim1 = t1
  g1[3] = (int)(((t1 >> 16) & 0xFFFFu) | ((t0 & 0xFFFFu) << 16));// tile_dim0
  g1[4] = (int)((t1 & 0xFFFFu) | ((t2 & 0xFFFFu) << 16));        // tile_dim1, tile_dim2
  g1[5] = (int)(uint32_t)(s0 & 0xFFFFFFFFull);                   // tensor_dim0_stride lo
  g1[6] = (int)(uint32_t)(((s0 >> 32) & 0xFFFFull) | ((s1 & 0xFFFFull) << 16));
  g1[7] = (int)(uint32_t)((s1 >> 16) & 0xFFFFFFFFull);           // tensor_dim1_stride hi
  i32x4 g2 = { (int)t2, 0, 0, 0 };                               // tensor_dim2 (0 if 2D)
  i32x4 g3 = { 0, 0, 0, 0 };
  i32x8 gz8 = { 0, 0, 0, 0, 0, 0, 0, 0 };
  __builtin_amdgcn_tensor_load_to_lds(g0, g1, g2, g3, gz8, 0);
}
#endif

// ================= K0: fold BN into weights, convert to bf16 =================
__global__ void k_fold_w1(const float* w1, const float* g1, const float* b1,
                          const float* m1, const float* v1,
                          bf16_t* w1b, float* bias1){
  int e = blockIdx.x * blockDim.x + threadIdx.x;
  if (e < 64*512){
    int oc = e >> 9;
    float s = g1[oc] * rsqrtf(v1[oc] + 1e-5f);
    w1b[e] = (bf16_t)(w1[e] * s);
  }
  if (e < 64){
    float s = g1[e] * rsqrtf(v1[e] + 1e-5f);
    bias1[e] = b1[e] - m1[e] * s;
  }
}

__global__ void k_fold_w2(const float* w2, const float* g2, const float* b2,
                          const float* m2, const float* v2,
                          bf16_t* w2b, float* bias2){
  int e = blockIdx.x * blockDim.x + threadIdx.x;
  if (e < C2*KTOT){
    int oc  = e / KTOT;
    int rem = e - oc * KTOT;
    int tap = rem / CCAT;
    int c   = rem - tap * CCAT;
    float s = g2[oc] * rsqrtf(v2[oc] + 1e-5f);
    // src OIHW: w2[((oc*1280 + c)*3 + kh)*3 + kw], tap = kh*3+kw
    w2b[e] = (bf16_t)(w2[(oc*CCAT + c)*9 + tap] * s);
  }
  if (e < C2){
    float s = g2[e] * rsqrtf(v2[e] + 1e-5f);
    bias2[e] = b2[e] - m2[e] * s;
  }
}

__global__ void k_fold_cw(const float* rw, bf16_t* cwb, float* cbb){
  int e = blockIdx.x * blockDim.x + threadIdx.x;
  if (e < M3P*1024){
    int m = e >> 10, c = e & 1023;
    cwb[e] = (m < M3) ? (bf16_t)rw[m*1025 + c] : (bf16_t)0.f;
  }
  if (e < M3P) cbb[e] = (e < M3) ? rw[e*1025 + 1024] : 0.f;
}

// ======= K1b: feat0 -> catp channels [256,1280) + zero the padded border =======
__global__ void k_fill_cat(const float* feat0, bf16_t* catp){
  int e = blockIdx.x * blockDim.x + threadIdx.x;
  if (e >= CCAT * NP) return;
  int c = e / NP;
  int p = e - c * NP;
  int b = p / NPPI;
  int r = p - b * NPPI;
  int y = r / HP, x = r - y * HP;
  bool interior = (y >= 1 && y <= HIN && x >= 1 && x <= HIN);
  if (c < 256){
    if (!interior) catp[e] = (bf16_t)0.f;     // interior written by k_conv1
  } else {
    float v = interior ? feat0[((b*1024 + (c-256))*HIN + (y-1))*HIN + (x-1)] : 0.f;
    catp[e] = (bf16_t)v;
  }
}

// ======= K1: 1x1 conv (512->64) + BN + leaky + reorg(2) -> catp channels [0,256) =======
__global__ __launch_bounds__(128) void k_conv1(const float* feat1, const bf16_t* w1b,
                                               const float* bias1, bf16_t* catp){
  __shared__ alignas(16) bf16_t sB[16 * 512];   // [px][c], 16 KB
  int tile = blockIdx.x;          // 23 tiles of 16 pixels over 361
  int shsw = blockIdx.y;          // 0..3
  int b    = blockIdx.z;          // 0..63
  int sh = shsw >> 1, sw = shsw & 1;
  int t = threadIdx.x;
  int n0 = tile * 16;
  for (int e = t; e < 16*512; e += 128){
    int px = e & 15;
    int c  = e >> 4;
    int n  = n0 + px; if (n > 360) n = 360;
    int h = n / 19, w = n - h * 19;
    int y1 = 2*h + sh, x1 = 2*w + sw;
    sB[px*512 + c] = (bf16_t)feat1[((b*512 + c)*38 + y1)*38 + x1];
  }
  __syncthreads();
  int wid = t >> 5, lane = t & 31;
  int c1 = wid * 16;
  f32x8 acc = {0.f,0.f,0.f,0.f,0.f,0.f,0.f,0.f};
#pragma unroll 1
  for (int ks = 0; ks < 512; ks += 32){
    bf16x16 a = load_a_frag(w1b + c1*512 + ks, 512, lane);
    bf16x16 bb = load_b_frag(sB + ks, 512, lane);
    acc = wmma_bf16(a, bb, acc);
  }
  int n = n0 + (lane & 15);
  if (n <= 360){
    int h = n / 19, w = n - h * 19;
    int p = b*NPPI + (h+1)*HP + (w+1);
    int mb = (lane >> 4) << 3;
#pragma unroll
    for (int r0 = 0; r0 < 8; ++r0){
      int m = c1 + r0 + mb;                     // conv1 output channel
      int k = shsw * 64 + m;                    // reorg channel
      float y = leaky(acc[r0] + bias1[m]);
      catp[(size_t)k * NP + p] = (bf16_t)y;
    }
  }
}

// ======= K2: 3x3 conv (1280->1024) as implicit GEMM, bf16 WMMA =======
// block = 256 thr (8 waves); tile 128 oc x 32 px.
// Per 32-channel step: ONE 3D TDM (32 c x 9 taps x 128 oc = 72 KB) + B stage,
// then 9 unrolled taps = 18 back-to-back WMMAs per wave between barriers.
#define SA2_BYTES (128*9*32*2)   // 73728
#define SB2_BYTES (76*32*2)      // 4864
__global__ __launch_bounds__(256) void k_conv2(const bf16_t* catp, const bf16_t* w2b,
                                               const float* bias2, bf16_t* prep){
  extern __shared__ char smem[];
  bf16_t* sA = (bf16_t*)smem;                    // [oc 128][tap 9][c 32] at LDS offset 0
  bf16_t* sB = (bf16_t*)(smem + SA2_BYTES);      // [px 76][c 32] transposed cat tile
  int nt = blockIdx.x;                           // 882 pixel tiles
  int mt = blockIdx.y;                           // 8 oc tiles
  int pbase = nt * 32;
  int oc0   = mt * 128;
  int t = threadIdx.x, wid = t >> 5, lane = t & 31;
  f32x8 acc0 = {0.f,0.f,0.f,0.f,0.f,0.f,0.f,0.f};
  f32x8 acc1 = {0.f,0.f,0.f,0.f,0.f,0.f,0.f,0.f};
#pragma unroll 1
  for (int cs = 0; cs < CCAT; cs += 32){
    __syncthreads();
    // stage B: 76 px x 32 c, transposed [px][c]; covers all 9 tap offsets (-22..+22)
    for (int e = t; e < 76*32; e += 256){
      int c  = e / 76;
      int px = e - c * 76;
      int pg = pbase - 22 + px;
      pg = pg < 0 ? 0 : (pg >= NP ? NP-1 : pg);
      sB[px*32 + c] = catp[(size_t)(cs + c) * NP + pg];
    }
    // stage A: 3D TDM tile (x=32 c, y=9 taps stride 1280, z=128 oc stride 11520)
#if defined(HAVE_TDM)
    if (t < 32){
      tdm_load_tile(0u, w2b + (size_t)oc0 * KTOT + cs,
                    32u, 9u, 128u, (uint64_t)CCAT, (uint64_t)KTOT);
#if defined(HAVE_TCNT)
      __builtin_amdgcn_s_wait_tensorcnt(0);
#endif
    }
#else
    for (int e = t; e < 128*9*32; e += 256){
      int oc = e / 288, rem = e - oc * 288;
      sA[e] = w2b[(size_t)(oc0 + oc) * KTOT + (rem >> 5) * CCAT + cs + (rem & 31)];
    }
#endif
    __syncthreads();
#pragma unroll
    for (int tap = 0; tap < 9; ++tap){
      int dy = tap/3 - 1, dx = tap - (tap/3)*3 - 1;
      int toff = 22 + dy*HP + dx;              // 0..44
      bf16x16 a  = load_a_frag(sA + (wid*16)*288 + tap*32, 288, lane);
      bf16x16 b0 = load_b_frag(sB + toff*32, 32, lane);
      bf16x16 b1 = load_b_frag(sB + (toff + 16)*32, 32, lane);
      acc0 = wmma_bf16(a, b0, acc0);
      acc1 = wmma_bf16(a, b1, acc1);
    }
  }
  // bias + leaky, store bf16 channel-major over padded pixels
  int n = lane & 15, mb = (lane >> 4) << 3;
#pragma unroll
  for (int r0 = 0; r0 < 8; ++r0){
    int oc = oc0 + wid*16 + r0 + mb;
    float bs = bias2[oc];
    prep[(size_t)oc * NP + pbase + n]      = (bf16_t)leaky(acc0[r0] + bs);
    prep[(size_t)oc * NP + pbase + 16 + n] = (bf16_t)leaky(acc1[r0] + bs);
  }
}

// ======= K3: einsum bchw,koc->bkohw (M=600->640, K=1024, N=28224), K-chunk 64 =======
#define SA3_BYTES (128*64*2)   // 16384
#define SB3_BYTES (32*64*2)    // 4096
__global__ __launch_bounds__(256) void k_einsum(const bf16_t* prep, const bf16_t* cwb,
                                                const float* cbb, float* out){
  extern __shared__ char smem[];
  bf16_t* sA = (bf16_t*)smem;                    // [m 128][c 64] at LDS offset 0
  bf16_t* sB = (bf16_t*)(smem + SA3_BYTES);      // [px 32][c 64]
  int nt = blockIdx.x;                           // 882
  int mt = blockIdx.y;                           // 5
  int pbase = nt * 32;
  int m0    = mt * 128;
  int t = threadIdx.x, wid = t >> 5, lane = t & 31;
  f32x8 acc0 = {0.f,0.f,0.f,0.f,0.f,0.f,0.f,0.f};
  f32x8 acc1 = {0.f,0.f,0.f,0.f,0.f,0.f,0.f,0.f};
#pragma unroll 1
  for (int cs = 0; cs < 1024; cs += 64){
    __syncthreads();
    for (int e = t; e < 32*64; e += 256){
      int px = e & 31, c = e >> 5;
      sB[px*64 + c] = prep[(size_t)(cs + c) * NP + pbase + px];
    }
#if defined(HAVE_TDM)
    if (t < 32){
      tdm_load_tile(0u, cwb + (size_t)m0 * 1024 + cs, 64u, 128u, 0u, 1024ull, 0ull);
#if defined(HAVE_TCNT)
      __builtin_amdgcn_s_wait_tensorcnt(0);
#endif
    }
#else
    for (int e = t; e < 128*64; e += 256){
      sA[e] = cwb[(size_t)(m0 + (e >> 6)) * 1024 + cs + (e & 63)];
    }
#endif
    __syncthreads();
#pragma unroll
    for (int kk = 0; kk < 64; kk += 32){
      bf16x16 a  = load_a_frag(sA + (wid*16)*64 + kk, 64, lane);
      bf16x16 b0 = load_b_frag(sB + kk, 64, lane);
      bf16x16 b1 = load_b_frag(sB + 16*64 + kk, 64, lane);
      acc0 = wmma_bf16(a, b0, acc0);
      acc1 = wmma_bf16(a, b1, acc1);
    }
  }
  // scatter valid pixels to output, add cb
  int n = lane & 15, mb = (lane >> 4) << 3;
#pragma unroll
  for (int g = 0; g < 2; ++g){
    int p = pbase + g*16 + n;
    int b = p / NPPI;
    int r = p - b * NPPI;
    int y = r / HP, x = r - y * HP;
    if (y >= 1 && y <= HIN && x >= 1 && x <= HIN){
      int h = y - 1, w = x - 1;
#pragma unroll
      for (int r0 = 0; r0 < 8; ++r0){
        int m = m0 + wid*16 + r0 + mb;
        if (m < M3){
          int k = m / 30, o = m - k * 30;
          float v = (g == 0 ? acc0[r0] : acc1[r0]) + cbb[m];
          out[(((b*20 + k)*30 + o)*361) + h*19 + w] = v;
        }
      }
    }
  }
}

// ================= host launcher =================
extern "C" void kernel_launch(void* const* d_in, const int* in_sizes, int n_in,
                              void* d_out, int out_size, void* d_ws, size_t ws_size,
                              hipStream_t stream){
  (void)in_sizes; (void)n_in; (void)out_size; (void)ws_size;
  const float* feat0 = (const float*)d_in[0];
  const float* feat1 = (const float*)d_in[1];
  const float* w1 = (const float*)d_in[2];
  const float* g1 = (const float*)d_in[3];
  const float* b1 = (const float*)d_in[4];
  const float* m1 = (const float*)d_in[5];
  const float* v1 = (const float*)d_in[6];
  const float* w2 = (const float*)d_in[7];
  const float* g2 = (const float*)d_in[8];
  const float* b2 = (const float*)d_in[9];
  const float* m2 = (const float*)d_in[10];
  const float* v2 = (const float*)d_in[11];
  const float* rw = (const float*)d_in[12];
  float* out = (float*)d_out;
  char* ws = (char*)d_ws;

  bf16_t* catp  = (bf16_t*)(ws + OFF_CATP);
  bf16_t* prep  = (bf16_t*)(ws + OFF_PREP);
  bf16_t* w1b   = (bf16_t*)(ws + OFF_W1B);
  bf16_t* w2b   = (bf16_t*)(ws + OFF_W2B);
  bf16_t* cwb   = (bf16_t*)(ws + OFF_CWB);
  float*  bias1 = (float*)(ws + OFF_B1);
  float*  bias2 = (float*)(ws + OFF_B2);
  float*  cbb   = (float*)(ws + OFF_CB);

  // conv2 needs 76.8 KB dynamic LDS (WGP has 320 KB); raise the per-kernel cap.
  (void)hipFuncSetAttribute((const void*)k_conv2,
                            hipFuncAttributeMaxDynamicSharedMemorySize,
                            SA2_BYTES + SB2_BYTES);

  k_fold_w1<<<(64*512 + 255)/256, 256, 0, stream>>>(w1, g1, b1, m1, v1, w1b, bias1);
  k_fold_w2<<<(C2*KTOT + 255)/256, 256, 0, stream>>>(w2, g2, b2, m2, v2, w2b, bias2);
  k_fold_cw<<<(M3P*1024 + 255)/256, 256, 0, stream>>>(rw, cwb, cbb);
  k_fill_cat<<<(CCAT*NP + 255)/256, 256, 0, stream>>>(feat0, catp);
  k_conv1<<<dim3(23, 4, BATCH), 128, 0, stream>>>(feat1, w1b, bias1, catp);
  k_conv2<<<dim3(NP/32, C2/128), 256, SA2_BYTES + SB2_BYTES, stream>>>(catp, w2b, bias2, prep);
  k_einsum<<<dim3(NP/32, M3P/128), 256, SA3_BYTES + SB3_BYTES, stream>>>(prep, cwb, cbb, out);
}
